// GNN_82626580840880
// MI455X (gfx1250) — matrix operand order repaired and checked by reading, not compiled, not run on hardware
//
#include <hip/hip_runtime.h>
#include <hip/hip_bf16.h>

// ---------- types for WMMA / TDM ----------
typedef __attribute__((ext_vector_type(8)))  __bf16 bf16x8;
typedef __attribute__((ext_vector_type(16))) __bf16 bf16x16;
typedef __attribute__((ext_vector_type(8)))  float  f32x8;
typedef __attribute__((ext_vector_type(4)))  unsigned int u32x4;
typedef __attribute__((ext_vector_type(4)))  int    i32x4;
typedef __attribute__((ext_vector_type(8)))  int    i32x8;

// ---------- bf16 helpers (bit-level, RNE) ----------
__device__ __forceinline__ unsigned short f2bf(float f) {
    unsigned int u = __float_as_uint(f);
    unsigned int r = u + 0x7FFFu + ((u >> 16) & 1u);
    return (unsigned short)(r >> 16);
}
__device__ __forceinline__ float bf2f(unsigned short s) {
    return __uint_as_float(((unsigned int)s) << 16);
}

// ---------- small elementwise kernels ----------
__global__ void cvt_f32_to_bf16_kernel(const float* __restrict__ src,
                                       unsigned short* __restrict__ dst,
                                       long long n4) {
    long long i = (long long)blockIdx.x * blockDim.x + threadIdx.x;
    if (i >= n4) return;
    float4 v = *(const float4*)(src + i * 4);
    ushort4 o;
    o.x = f2bf(v.x); o.y = f2bf(v.y); o.z = f2bf(v.z); o.w = f2bf(v.w);
    *(ushort4*)(dst + i * 4) = o;
}

// Wt[h*K + k] = bf16(W[k*H + h])   (transpose so B-fragment loads are contiguous)
__global__ void cvt_w_transpose_kernel(const float* __restrict__ W,
                                       unsigned short* __restrict__ Wt,
                                       int K, int H) {
    int i = blockIdx.x * blockDim.x + threadIdx.x;
    if (i >= K * H) return;
    int h = i / K, k = i % K;
    Wt[i] = f2bf(W[(size_t)k * H + h]);
}

__global__ void deg_init_kernel(float* __restrict__ deg, int N) {
    int i = blockIdx.x * blockDim.x + threadIdx.x;
    if (i < N) deg[i] = 1.0f;   // self-loop
}

__global__ void deg_scatter_kernel(const int* __restrict__ ei, float* __restrict__ deg, int E) {
    int e = blockIdx.x * blockDim.x + threadIdx.x;
    if (e < E) atomicAdd(&deg[ei[E + e]], 1.0f);   // dst = ei[1][e]
}

__global__ void dinv_kernel(const float* __restrict__ deg, float* __restrict__ dinv, int N) {
    int i = blockIdx.x * blockDim.x + threadIdx.x;
    if (i < N) dinv[i] = rsqrtf(deg[i]);
}

// ---------- WMMA GEMM: C[M x 256] = A[M x K] (bf16) @ Wt^T, Wt is [256 x K] bf16 ----------
template <int K>
__global__ __launch_bounds__(256) void gemm_bf16_wmma(
    const __bf16* __restrict__ A,
    const __bf16* __restrict__ Wt,
    float* __restrict__ C) {
    __shared__ __align__(16) __bf16 ldsW[64 * K];   // 64-column weight tile

    const int tid  = threadIdx.x;
    const int lane = tid & 31;
    const int wave = tid >> 5;                  // 8 waves: 2 (M) x 4 (N)
    const int colBlock = blockIdx.y * 64;

#if defined(__gfx1250__) && __has_builtin(__builtin_amdgcn_tensor_load_to_lds)
    // ---- Tensor Data Mover: DMA the [64 x K] bf16 weight tile into LDS ----
    if (wave == 0) {
        unsigned long long ga =
            (unsigned long long)(const void*)(Wt + (size_t)colBlock * K);
        unsigned lds_off = (unsigned)(unsigned long long)(void*)&ldsW[0];
        u32x4 g0;
        g0[0] = 1u;                                          // count=1 (valid user D#)
        g0[1] = lds_off;                                     // lds_addr (bytes)
        g0[2] = (unsigned)(ga & 0xFFFFFFFFull);              // global_addr[31:0]
        g0[3] = (unsigned)((ga >> 32) & 0x01FFFFFFull)       // global_addr[56:32]
                | (2u << 30);                                // type=2 ("image")
        i32x8 g1;
        g1[0] = 0x00010000;                                  // data_size=1 -> 2 bytes
        g1[1] = (K & 0xFFFF) << 16;                          // tensor_dim0[15:0]=K
        g1[2] = (64 << 16);                                  // tensor_dim1[15:0]=64 rows
        g1[3] = (K << 16);                                   // tile_dim0=K
        g1[4] = 64;                                          // tile_dim1=64, tile_dim2=0
        g1[5] = K;                                           // tensor_dim0_stride[31:0]=K
        g1[6] = 0;                                           // stride hi / dim1_stride lo
        g1[7] = 0;
        i32x4 gz = {0, 0, 0, 0};                             // groups 2/3: dims unused
#if __clang_major__ >= 23
        i32x8 gz8 = {0, 0, 0, 0, 0, 0, 0, 0};
        __builtin_amdgcn_tensor_load_to_lds(g0, g1, gz, gz, gz8, 0);
#else
        __builtin_amdgcn_tensor_load_to_lds(g0, g1, gz, gz, 0);
#endif
        __builtin_amdgcn_s_wait_tensorcnt(0);
    }
    __syncthreads();
#else
    // ---- fallback: cooperative vector load of the weight tile ----
    for (int i = tid; i < (64 * K) / 8; i += 256) {
        int c = (i * 8) / K;
        int k = (i * 8) % K;
        *(bf16x8*)&ldsW[c * K + k] = *(const bf16x8*)&Wt[(size_t)(colBlock + c) * K + k];
    }
    __syncthreads();
#endif

    const int rowBase = blockIdx.x * 32 + (wave >> 2) * 16;
    const int colTile = (wave & 3) * 16;        // within LDS tile
    const int m  = lane & 15;
    const int hi = lane >> 4;

    const __bf16* arow = A + (size_t)(rowBase + m) * K;
    const __bf16* brow = &ldsW[(colTile + m) * K];   // lane's B column (= row of Wt)

    f32x8 acc = {};
#pragma unroll
    for (int kb = 0; kb < K; kb += 32) {
        // A 16x32 fragment: lane (m,hi): e=0..7 -> k=kb+8*hi+e ; e=8..15 -> k=kb+16+8*hi+(e-8)
        bf16x8 a0 = *(const bf16x8*)&arow[kb + 8 * hi];
        bf16x8 a1 = *(const bf16x8*)&arow[kb + 16 + 8 * hi];
        bf16x16 a = __builtin_shufflevector(a0, a1, 0,1,2,3,4,5,6,7,8,9,10,11,12,13,14,15);
        // B 32x16 fragment: lane (n=m,hi): e=0..7 -> k=kb+8*hi+e ; e=8..15 -> k=kb+16+8*hi+(e-8)
        bf16x8 b0 = *(const bf16x8*)&brow[kb + 8 * hi];
        bf16x8 b1 = *(const bf16x8*)&brow[kb + 16 + 8 * hi];
        bf16x16 b = __builtin_shufflevector(b0, b1, 0,1,2,3,4,5,6,7,8,9,10,11,12,13,14,15);
        acc = __builtin_amdgcn_wmma_f32_16x16x32_bf16(false, a, false, b,
                                                      (short)0, acc, false, false);
    }

    // C/D layout: VGPR r -> row = rowBase + r + 8*hi, col = colBlock + colTile + (lane&15)
    float* crow = C + (size_t)(rowBase + 8 * hi) * 256 + colBlock + colTile + m;
#pragma unroll
    for (int r = 0; r < 8; ++r) crow[(size_t)r * 256] = acc[r];
}

// ---------- self-loop init: agg[b,n,:] = dinv[n]^2 * h[b,n,:] ----------
__global__ void init_agg_kernel(const float* __restrict__ h,
                                const float* __restrict__ dinv,
                                float* __restrict__ agg, int N, long long total4) {
    long long i = (long long)blockIdx.x * blockDim.x + threadIdx.x;
    if (i >= total4) return;
    int f4  = (int)(i & 63);
    long long bn = i >> 6;
    int n = (int)(bn % N);
    float d = dinv[n];
    float s = d * d;
    float4 v = *(const float4*)(h + bn * 256 + f4 * 4);
    float4 o; o.x = s * v.x; o.y = s * v.y; o.z = s * v.z; o.w = s * v.w;
    *(float4*)(agg + bn * 256 + f4 * 4) = o;
}

// ---------- edge scatter: agg[b,dst,:] += dinv[src]*dinv[dst] * h[b,src,:] ----------
__global__ void edge_scatter_kernel(const int* __restrict__ ei,
                                    const float* __restrict__ dinv,
                                    const float* __restrict__ h,
                                    float* __restrict__ agg,
                                    int E, int N, long long total) {
    long long i = (long long)blockIdx.x * blockDim.x + threadIdx.x;
    if (i >= total) return;
    int f4 = (int)(i & 63);
    long long eb = i >> 6;
    int e = (int)(eb % E);
    int b = (int)(eb / E);
    int s = ei[e];
    int d = ei[E + e];
    float nrm = dinv[s] * dinv[d];
    float4 v = *(const float4*)(h + ((size_t)b * N + s) * 256 + f4 * 4);
    float* ap = agg + ((size_t)b * N + d) * 256 + f4 * 4;
    atomicAdd(ap + 0, nrm * v.x);
    atomicAdd(ap + 1, nrm * v.y);
    atomicAdd(ap + 2, nrm * v.z);
    atomicAdd(ap + 3, nrm * v.w);
}

// ---------- bias + ReLU + convert to bf16 activations ----------
__global__ void bias_relu_cvt_kernel(const float* __restrict__ agg,
                                     const float* __restrict__ bias,
                                     unsigned short* __restrict__ act,
                                     long long total4) {
    long long i = (long long)blockIdx.x * blockDim.x + threadIdx.x;
    if (i >= total4) return;
    int f4 = (int)(i & 63);
    float4 v = *(const float4*)(agg + i * 4);
    const float4 bsv = *(const float4*)(bias + f4 * 4);
    ushort4 o;
    o.x = f2bf(fmaxf(v.x + bsv.x, 0.0f));
    o.y = f2bf(fmaxf(v.y + bsv.y, 0.0f));
    o.z = f2bf(fmaxf(v.z + bsv.z, 0.0f));
    o.w = f2bf(fmaxf(v.w + bsv.w, 0.0f));
    *(ushort4*)(act + i * 4) = o;
}

// ---------- head ----------
__global__ void out_init_kernel(const float* __restrict__ bl, float* __restrict__ out, int n) {
    int i = blockIdx.x * blockDim.x + threadIdx.x;
    if (i < n) out[i] = bl[0];
}

__global__ void head_kernel(const unsigned short* __restrict__ act,
                            const float* __restrict__ Wl,
                            float* __restrict__ out, int N) {
    __shared__ float red[256];
    int b = blockIdx.y;
    int n = blockIdx.x * blockDim.x + threadIdx.x;
    float acc = 0.0f;
    if (n < N) {
        const unsigned short* row = act + ((size_t)b * N + n) * 256;
        for (int f = 0; f < 256; ++f) acc += bf2f(row[f]) * Wl[f];
    }
    red[threadIdx.x] = acc;
    __syncthreads();
    for (int s = 128; s > 0; s >>= 1) {
        if (threadIdx.x < s) red[threadIdx.x] += red[threadIdx.x + s];
        __syncthreads();
    }
    if (threadIdx.x == 0) atomicAdd(&out[b], red[0] / (float)N);
}

// ---------- launch ----------
extern "C" void kernel_launch(void* const* d_in, const int* in_sizes, int n_in,
                              void* d_out, int out_size, void* d_ws, size_t ws_size,
                              hipStream_t stream) {
    const float* x  = (const float*)d_in[0];
    const int*   ei = (const int*)d_in[1];
    const float* W1 = (const float*)d_in[2];
    const float* b1 = (const float*)d_in[3];
    const float* W2 = (const float*)d_in[4];
    const float* b2 = (const float*)d_in[5];
    const float* W3 = (const float*)d_in[6];
    const float* b3 = (const float*)d_in[7];
    const float* Wl = (const float*)d_in[8];
    const float* bl = (const float*)d_in[9];

    const int B = 2, Fin = 128, H = 256;
    const int N = in_sizes[0] / (B * Fin);
    const int E = in_sizes[1] / 2;
    const long long M = (long long)B * N;            // 40000 rows (multiple of 32)

    // ---- workspace carve ----
    char* p = (char*)d_ws;
    auto carve = [&](size_t bytes) -> void* {
        void* r = (void*)p;
        p += (bytes + 255) & ~(size_t)255;
        return r;
    };
    float*          deg  = (float*)carve((size_t)N * 4);
    float*          dinv = (float*)carve((size_t)N * 4);
    unsigned short* xb   = (unsigned short*)carve((size_t)B * N * Fin * 2);
    unsigned short* Wt1  = (unsigned short*)carve((size_t)Fin * H * 2);
    unsigned short* Wt2  = (unsigned short*)carve((size_t)H * H * 2);
    unsigned short* Wt3  = (unsigned short*)carve((size_t)H * H * 2);
    float*          hbuf = (float*)carve((size_t)B * N * H * 4);
    float*          agg  = (float*)carve((size_t)B * N * H * 4);
    unsigned short* act  = (unsigned short*)carve((size_t)B * N * H * 2);
    float*          out  = (float*)d_out;

    const int TB = 256;
    auto blk = [](long long n, int t) { return (unsigned)((n + t - 1) / t); };

    // ---- precompute: conversions + norm ----
    long long x4 = (long long)B * N * Fin / 4;
    cvt_f32_to_bf16_kernel<<<blk(x4, TB), TB, 0, stream>>>(x, xb, x4);
    cvt_w_transpose_kernel<<<blk((long long)Fin * H, TB), TB, 0, stream>>>(W1, Wt1, Fin, H);
    cvt_w_transpose_kernel<<<blk((long long)H * H, TB), TB, 0, stream>>>(W2, Wt2, H, H);
    cvt_w_transpose_kernel<<<blk((long long)H * H, TB), TB, 0, stream>>>(W3, Wt3, H, H);
    deg_init_kernel<<<blk(N, TB), TB, 0, stream>>>(deg, N);
    deg_scatter_kernel<<<blk(E, TB), TB, 0, stream>>>(ei, deg, E);
    dinv_kernel<<<blk(N, TB), TB, 0, stream>>>(deg, dinv, N);
    out_init_kernel<<<1, 32, 0, stream>>>(bl, out, out_size);

    long long total4 = M * 64;                       // (B*N*256)/4
    long long etotal = (long long)E * B * 64;        // edges * batch * (256/4)
    dim3 ggrid((unsigned)(M / 32), 4);

    // ---- layer 1: K=128 ----
    gemm_bf16_wmma<128><<<ggrid, 256, 0, stream>>>((const __bf16*)xb, (const __bf16*)Wt1, hbuf);
    init_agg_kernel<<<blk(total4, TB), TB, 0, stream>>>(hbuf, dinv, agg, N, total4);
    edge_scatter_kernel<<<blk(etotal, TB), TB, 0, stream>>>(ei, dinv, hbuf, agg, E, N, etotal);
    bias_relu_cvt_kernel<<<blk(total4, TB), TB, 0, stream>>>(agg, b1, act, total4);

    // ---- layer 2: K=256 ----
    gemm_bf16_wmma<256><<<ggrid, 256, 0, stream>>>((const __bf16*)act, (const __bf16*)Wt2, hbuf);
    init_agg_kernel<<<blk(total4, TB), TB, 0, stream>>>(hbuf, dinv, agg, N, total4);
    edge_scatter_kernel<<<blk(etotal, TB), TB, 0, stream>>>(ei, dinv, hbuf, agg, E, N, etotal);
    bias_relu_cvt_kernel<<<blk(total4, TB), TB, 0, stream>>>(agg, b2, act, total4);

    // ---- layer 3: K=256 ----
    gemm_bf16_wmma<256><<<ggrid, 256, 0, stream>>>((const __bf16*)act, (const __bf16*)Wt3, hbuf);
    init_agg_kernel<<<blk(total4, TB), TB, 0, stream>>>(hbuf, dinv, agg, N, total4);
    edge_scatter_kernel<<<blk(etotal, TB), TB, 0, stream>>>(ei, dinv, hbuf, agg, E, N, etotal);
    bias_relu_cvt_kernel<<<blk(total4, TB), TB, 0, stream>>>(agg, b3, act, total4);

    // ---- mean-pool + linear head ----
    dim3 hgrid(blk(N, TB), B);
    head_kernel<<<hgrid, TB, 0, stream>>>(act, Wl, out, N);
}
